// QuantizedLinearInt8_1013612282098
// MI455X (gfx1250) — matrix operand order repaired and checked by reading, not compiled, not run on hardware
//
#include <hip/hip_runtime.h>
#include <stdint.h>

// out[M,N] = x[M,K](fp16) . dequant(W[N,K] int8, scale[N])^T + bias[N]
// M = 8192, K = 4096, N = 11008.  AI ~2550 FLOP/B -> compute bound on
// v_wmma_f32_16x16x32_f16.  W is dequantized once into d_ws (int8->fp16 is
// exact; ~6us of HBM traffic), then the GEMM inner loop is pure WMMA +
// ds_load_b128, with staging done by GLOBAL_LOAD_ASYNC_TO_LDS_B128 (ASYNCcnt)
// so no VGPR round-trip and no loadcnt stalls inside a stage.

typedef _Float16 v16h __attribute__((ext_vector_type(16)));
typedef _Float16 v8h  __attribute__((ext_vector_type(8)));
typedef _Float16 v2h  __attribute__((ext_vector_type(2)));
typedef float    v8f  __attribute__((ext_vector_type(8)));

constexpr int M = 4 * 2048;   // B*S
constexpr int K = 4096;       // D_IN
constexpr int N = 11008;      // D_OUT

constexpr int BM = 128, BN = 256, BK = 32;
constexpr int BKP = BK + 8;          // padded LDS row (80B stride, 16B aligned)
constexpr int NSTAGE = K / BK;       // 128

union V16 { v16h v; v8h h[2]; uint32_t u[8]; };

// ---- CDNA5 async global->LDS copy (tracked by ASYNCcnt) -------------------
__device__ __forceinline__ uint32_t lds_off(const void* p) {
    // generic -> addrspace(3) cast folds to the raw LDS byte offset, which is
    // what the VDST operand of global_load_async_to_lds consumes.
    return (uint32_t)(uintptr_t)(__attribute__((address_space(3))) const void*)p;
}

template <int OFF>
__device__ __forceinline__ void async_ld_b128(uint32_t lds, const void* gaddr) {
    // IOFFSET is added to BOTH the LDS and the global address (ISA 10.x).
    asm volatile("global_load_async_to_lds_b128 %0, %1, off offset:%2"
                 :: "v"(lds), "v"(gaddr), "n"(OFF) : "memory");
}

__device__ __forceinline__ void wait_async0() {
    asm volatile("s_wait_asynccnt 0x0" ::: "memory");
}

// ---- exact packed int8 -> fp16 (magic exponent trick) ---------------------
__device__ __forceinline__ void cvt4(uint32_t w, uint32_t& o0, uint32_t& o1) {
    uint32_t lo = __builtin_amdgcn_perm(0u, w, 0x0c010c00u) ^ 0x64806480u;
    uint32_t hi = __builtin_amdgcn_perm(0u, w, 0x0c030c02u) ^ 0x64806480u;
    const v2h bias1152 = {(_Float16)1152, (_Float16)1152};
    v2h a = __builtin_bit_cast(v2h, lo) - bias1152;
    v2h b = __builtin_bit_cast(v2h, hi) - bias1152;
    o0 = __builtin_bit_cast(uint32_t, a);
    o1 = __builtin_bit_cast(uint32_t, b);
}

__device__ __forceinline__ V16 dequant16(uint4 q) {
    V16 r;
    cvt4(q.x, r.u[0], r.u[1]);
    cvt4(q.y, r.u[2], r.u[3]);
    cvt4(q.z, r.u[4], r.u[5]);
    cvt4(q.w, r.u[6], r.u[7]);
    return r;
}

// ---------------- W dequant prepass: 45MB int8 -> 90MB fp16 ----------------
__global__ __launch_bounds__(256)
void dequant_w_kernel(const int8_t* __restrict__ wq, _Float16* __restrict__ wf) {
    const size_t idx = ((size_t)blockIdx.x * 256 + threadIdx.x) * 16;
    uint4 q = *(const uint4*)(wq + idx);
    V16 h = dequant16(q);
    *(v8h*)(wf + idx)     = h.h[0];
    *(v8h*)(wf + idx + 8) = h.h[1];
}

// ---------------- main GEMM ----------------
template <bool B_FP16>
__global__ __launch_bounds__(256)
void qlinear_gemm_kernel(const _Float16* __restrict__ x,
                         const void* __restrict__ wvoid,
                         const _Float16* __restrict__ scale,
                         const _Float16* __restrict__ bias,
                         _Float16* __restrict__ out)
{
    __shared__ _Float16 As[2][BM][BKP];   // 20 KB
    __shared__ _Float16 Bs[2][BN][BKP];   // 40 KB

    const int tid  = threadIdx.x;
    const int lane = tid & 31;
    const int wave = tid >> 5;
    const int wm   = wave & 1;            // 2 waves along M
    const int wn   = wave >> 1;           // 4 waves along N
    const int r    = lane & 15;
    const int hi   = lane >> 4;

    const int blockM = blockIdx.y * BM;
    const int blockN = blockIdx.x * BN;

    // A staging: 2 threads/row, 16 halves (32B) each.  B: 1 thread/row, 64B.
    const int arow = tid >> 1;
    const int achk = (tid & 1) * 16;
    const _Float16* gA  = x + (size_t)(blockM + arow) * K + achk;
    const _Float16* gBf = (const _Float16*)wvoid + (size_t)(blockN + tid) * K;
    const int8_t*   gBq = (const int8_t*)  wvoid + (size_t)(blockN + tid) * K;

    const uint32_t ldsA[2] = { lds_off(&As[0][arow][achk]), lds_off(&As[1][arow][achk]) };
    const uint32_t ldsB[2] = { lds_off(&Bs[0][tid][0]),     lds_off(&Bs[1][tid][0]) };

    v8f acc[4][4] = {};

    // ---- stage one k-tile into a buffer ----
    auto stage = [&](int nbuf, int k0) {
        if (B_FP16) {
            async_ld_b128<0> (ldsA[nbuf], gA + k0);
            async_ld_b128<16>(ldsA[nbuf], gA + k0);
            async_ld_b128<0> (ldsB[nbuf], gBf + k0);
            async_ld_b128<16>(ldsB[nbuf], gBf + k0);
            async_ld_b128<32>(ldsB[nbuf], gBf + k0);
            async_ld_b128<48>(ldsB[nbuf], gBf + k0);
        } else {
            async_ld_b128<0> (ldsA[nbuf], gA + k0);
            async_ld_b128<16>(ldsA[nbuf], gA + k0);
            V16 h0 = dequant16(*(const uint4*)(gBq + k0));
            V16 h1 = dequant16(*(const uint4*)(gBq + k0 + 16));
            *(v8h*)&Bs[nbuf][tid][0]  = h0.h[0];
            *(v8h*)&Bs[nbuf][tid][8]  = h0.h[1];
            *(v8h*)&Bs[nbuf][tid][16] = h1.h[0];
            *(v8h*)&Bs[nbuf][tid][24] = h1.h[1];
        }
    };

    // prologue: k-tile 0 -> buffer 0
    stage(0, 0);
    wait_async0();
    __syncthreads();

    for (int kt = 0; kt < NSTAGE; ++kt) {
        const int buf = kt & 1;

        // issue next stage's async copies before the math
        if (kt + 1 < NSTAGE) {
            stage(buf ^ 1, (kt + 1) * BK);
            if (kt + 2 < NSTAGE) {   // warm near caches for the stage after next
                __builtin_prefetch(gA + (size_t)(kt + 2) * BK, 0, 3);
                if (B_FP16) __builtin_prefetch(gBf + (size_t)(kt + 2) * BK, 0, 3);
                else        __builtin_prefetch(gBq + (size_t)(kt + 2) * BK, 0, 3);
            }
        }

        // A fragments (16x32 MxK): lanes 0-15 K=0..7 & 16..23, lanes 16-31 K=8..15 & 24..31
        V16 afrag[4];
#pragma unroll
        for (int mi = 0; mi < 4; ++mi) {
            const int row = wm * 64 + mi * 16 + r;
            afrag[mi].h[0] = *(const v8h*)&As[buf][row][hi * 8];
            afrag[mi].h[1] = *(const v8h*)&As[buf][row][hi * 8 + 16];
        }
        // B fragments (32x16 KxN): lanes 0-15 K=0..7 & 16..23, lanes 16-31 K=8..15 & 24..31
#pragma unroll
        for (int ni = 0; ni < 4; ++ni) {
            const int row = wn * 64 + ni * 16 + r;
            V16 bfrag;
            bfrag.h[0] = *(const v8h*)&Bs[buf][row][hi * 8];
            bfrag.h[1] = *(const v8h*)&Bs[buf][row][hi * 8 + 16];
#pragma unroll
            for (int mi = 0; mi < 4; ++mi) {
                acc[mi][ni] = __builtin_amdgcn_wmma_f32_16x16x32_f16(
                    false, afrag[mi].v, false, bfrag.v,
                    (short)0, acc[mi][ni], false, false);
            }
        }

        // next buffer must be fully in LDS before anyone crosses the barrier
        wait_async0();
        __syncthreads();
    }

    // epilogue: fold per-row scale + bias, store fp16
#pragma unroll
    for (int ni = 0; ni < 4; ++ni) {
        const int n  = blockN + wn * 64 + ni * 16 + r;
        const float s  = (float)scale[n];
        const float bb = (float)bias[n];
#pragma unroll
        for (int mi = 0; mi < 4; ++mi) {
            const int m0 = blockM + wm * 64 + mi * 16 + hi * 8;
#pragma unroll
            for (int v = 0; v < 8; ++v)
                out[(size_t)(m0 + v) * N + n] = (_Float16)(acc[mi][ni][v] * s + bb);
        }
    }
}

extern "C" void kernel_launch(void* const* d_in, const int* in_sizes, int n_in,
                              void* d_out, int out_size, void* d_ws, size_t ws_size,
                              hipStream_t stream) {
    const _Float16* x   = (const _Float16*)d_in[0];
    const int8_t*   wq  = (const int8_t*)d_in[1];
    const _Float16* scl = (const _Float16*)d_in[2];
    const _Float16* bia = (const _Float16*)d_in[3];
    _Float16*       out = (_Float16*)d_out;

    dim3 grid(N / BN, M / BM);                     // 43 x 64 tiles
    const size_t wbytes = (size_t)N * K * sizeof(_Float16);
    if (ws_size >= wbytes) {
        const unsigned nblk = (unsigned)(((size_t)N * K) / (16 * 256));  // 11008
        dequant_w_kernel<<<dim3(nblk), dim3(256), 0, stream>>>(wq, (_Float16*)d_ws);
        qlinear_gemm_kernel<true><<<grid, dim3(256), 0, stream>>>(x, d_ws, scl, bia, out);
    } else {
        qlinear_gemm_kernel<false><<<grid, dim3(256), 0, stream>>>(x, wq, scl, bia, out);
    }
}
